// SwinBlock_38843684225386
// MI455X (gfx1250) — matrix operand order repaired and checked by reading, not compiled, not run on hardware
//
#include <hip/hip_runtime.h>
#include <hip/hip_bf16.h>

typedef _Float16 half_t;
typedef __attribute__((ext_vector_type(16))) _Float16 v16h;
typedef __attribute__((ext_vector_type(8)))  _Float16 v8h;
typedef __attribute__((ext_vector_type(8)))  float    v8f;

// ---------------- constants ----------------
#define CDIM   768
#define HEADS  16
#define CH     48
#define LWIN   64      // 8*8 tokens per window
#define MROWS  32768   // 8*8*8*64 rows in window order
#define MODSTRIDE 4608 // 6*C

// workspace offsets (bytes), all 256-aligned
#define OFF_M     ((size_t)0)                    // 8*4608*4      = 147456
#define OFF_CTAB  ((size_t)147456)               // 2*64*384*4    = 196608
#define OFF_WQ    ((size_t)344064)               // 768*2304*2    = 3538944
#define OFF_WP    ((size_t)3883008)              // 768*768*2     = 1179648
#define OFF_W1    ((size_t)5062656)              // 768*3072*2    = 4718592
#define OFF_W2    ((size_t)9781248)              // 3072*768*2    = 4718592
#define OFF_Y     ((size_t)14499840)             // 32768*768*2   = 50331648 (reused as Z)
#define OFF_QKV   ((size_t)64831488)             // max(QKV 151MB, H 201MB) = 201326592
#define OFF_O     ((size_t)266158080)            // 32768*768*2   = 50331648
#define OFF_Y1    ((size_t)316489728)            // 32768*768*4   = 100663296

// ---------------- async global->LDS (CDNA5), guarded fallback ----------------
#if defined(__gfx1250__) && __has_builtin(__builtin_amdgcn_global_load_async_to_lds_b128)
#define HAVE_ASYNC_LDS 1
#endif

typedef int v4i_vs __attribute__((vector_size(16)));
typedef __attribute__((address_space(1))) v4i_vs* as1_v4i_ptr;
typedef __attribute__((address_space(3))) v4i_vs* as3_v4i_ptr;

__device__ inline void async_copy16(const void* g, void* l) {
#ifdef HAVE_ASYNC_LDS
    __builtin_amdgcn_global_load_async_to_lds_b128(
        (as1_v4i_ptr)(uintptr_t)g,
        (as3_v4i_ptr)(uintptr_t)l, 0, 0);
#else
    *(uint4*)l = *(const uint4*)g;
#endif
}

__device__ inline void async_wait0() {
#ifdef HAVE_ASYNC_LDS
#if __has_builtin(__builtin_amdgcn_s_wait_asynccnt)
    __builtin_amdgcn_s_wait_asynccnt(0);
#endif
#endif
}

// ---------------- WMMA fragment loaders (16x16x32 f16) ----------------
// A (16x32, row-major source, stride lda halves):
//   lanes 0-15 : M=lane,    h[0..15]=K(kb..kb+15), kb=0
//   lanes16-31 : M=lane-16, same but kb=16
__device__ inline v16h frag_a(const half_t* A, int lda, int m0, int k0, int lane) {
    int m  = m0 + (lane & 15);
    int kb = k0 + ((lane >> 4) << 4);
    const half_t* p = A + (size_t)m * lda + kb;
    v16h r;
#pragma unroll
    for (int i = 0; i < 16; ++i) r[i] = p[i];
    return r;
}
// B (32x16) loaded from a K-transposed buffer Bt where Bt[n][k] = B[k][n]:
//   lanes 0-15 : N=lane,    h[i]=B[k0+i][N]      (i=0..15)
//   lanes16-31 : N=lane-16, h[i]=B[k0+16+i][N]
__device__ inline v16h frag_bt(const half_t* Bt, int ldb, int k0, int n0, int lane) {
    int n  = n0 + (lane & 15);
    int kb = k0 + ((lane >> 4) << 4);
    const half_t* p = Bt + (size_t)n * ldb + kb;
    v16h r;
#pragma unroll
    for (int i = 0; i < 16; ++i) r[i] = p[i];
    return r;
}

__device__ inline v8f wmma_f16(v16h a, v16h b, v8f c) {
    return __builtin_amdgcn_wmma_f32_16x16x32_f16(false, a, false, b, (short)0, c, false, false);
}

// ---------------- tiny kernels ----------------
__global__ __launch_bounds__(256) void f32_to_f16_kernel(const float* __restrict__ in,
                                                         half_t* __restrict__ out, int n) {
    int i = blockIdx.x * 256 + threadIdx.x;
    if (i < n) out[i] = (half_t)in[i];
}

__global__ __launch_bounds__(256) void rope_table_kernel(const float* __restrict__ theta,
                                                         float* __restrict__ ctab) {
    int i = blockIdx.x * 256 + threadIdx.x;
    if (i >= 64 * 384) return;
    int l = i / 384, j = i - l * 384;
    float th = (float)(l >> 3) * theta[j] + (float)(l & 7) * theta[384 + j];
    ctab[i] = cosf(th);
    ctab[24576 + i] = sinf(th);
}

// m = silu(mod@w1+b1)@w2 + b2, per batch row
__global__ __launch_bounds__(256) void ada_kernel(const float* __restrict__ mod,
                                                  const float* __restrict__ w1,
                                                  const float* __restrict__ b1,
                                                  const float* __restrict__ w2,
                                                  const float* __restrict__ b2,
                                                  float* __restrict__ mout) {
    __shared__ float xm[768];
    __shared__ float h[768];
    int b = blockIdx.x, tid = threadIdx.x;
    for (int c = tid; c < 768; c += 256) xm[c] = mod[b * 768 + c];
    __syncthreads();
    for (int c = tid; c < 768; c += 256) {
        float dot = 0.f;
        for (int k = 0; k < 768; ++k) dot += xm[k] * w1[k * 768 + c];
        dot += b1[c];
        h[c] = dot / (1.0f + __expf(-dot));
    }
    __syncthreads();
    for (int c = tid; c < MODSTRIDE; c += 256) {
        float dot = 0.f;
        for (int k = 0; k < 768; ++k) dot += h[k] * w2[k * MODSTRIDE + c];
        mout[b * MODSTRIDE + c] = dot + b2[c];
    }
}

// LayerNorm(ddof=1) + (a+1)*ln + b ; MODE 0: x -> shifted/window order Y ; MODE 1: y1 -> Z (plain)
template <int MODE>
__global__ __launch_bounds__(256) void ln_mod_kernel(const float* __restrict__ X,
                                                     const float* __restrict__ m,
                                                     half_t* __restrict__ Yout) {
    __shared__ float red[256];
    int r = blockIdx.x, tid = threadIdx.x;
    int b = r >> 12;
    size_t srcbase;
    if constexpr (MODE == 0) {
        int rem = r & 4095, win = rem >> 6, l = rem & 63;
        int hs  = ((win >> 3) << 3) + (l >> 3);
        int wsx = ((win & 7) << 3) + (l & 7);
        int hh = (hs + 60) & 63, ww = (wsx + 60) & 63;
        srcbase = (((size_t)b << 12) + (hh << 6) + ww) * CDIM;
    } else {
        srcbase = (size_t)r * CDIM;
    }
    float x0 = X[srcbase + tid];
    float x1 = X[srcbase + tid + 256];
    float x2 = X[srcbase + tid + 512];
    red[tid] = x0 + x1 + x2;
    __syncthreads();
    for (int off = 128; off > 0; off >>= 1) {
        if (tid < off) red[tid] += red[tid + off];
        __syncthreads();
    }
    float mean = red[0] * (1.0f / 768.0f);
    __syncthreads();
    float d0 = x0 - mean, d1 = x1 - mean, d2 = x2 - mean;
    red[tid] = d0 * d0 + d1 * d1 + d2 * d2;
    __syncthreads();
    for (int off = 128; off > 0; off >>= 1) {
        if (tid < off) red[tid] += red[tid + off];
        __syncthreads();
    }
    float rs = rsqrtf(red[0] * (1.0f / 767.0f) + 1e-5f);
    const int aoff = (MODE == 0) ? 0 : 2304;
    const int boff = (MODE == 0) ? 768 : 3072;
    const float* mb = m + (size_t)b * MODSTRIDE;
#pragma unroll
    for (int t = 0; t < 3; ++t) {
        int c = tid + t * 256;
        float xv = (t == 0) ? d0 : (t == 1) ? d1 : d2;
        float o = (mb[aoff + c] + 1.0f) * (xv * rs) + mb[boff + c];
        Yout[(size_t)r * CDIM + c] = (half_t)o;
    }
}

// ---------------- tiled WMMA GEMM with fused epilogues ----------------
// EPI 0: QKV  (bias + RoPE on q/k, f16 out, N=2304)
// EPI 1: PROJ (bias + un-window/un-shift + c1 residual vs x -> f32 y1)
// EPI 2: MLP1 (bias + SiLU, f16 out)
// EPI 3: MLP2 (bias + c2 residual vs x -> f32 d_out)
template <int EPI>
__global__ __launch_bounds__(128) void gemm_kernel(const half_t* __restrict__ A,
                                                   const half_t* __restrict__ B,
                                                   const float* __restrict__ bias,
                                                   int M, int N, int K,
                                                   const float* __restrict__ mmod,
                                                   const float* __restrict__ ctab,
                                                   const float* __restrict__ xres,
                                                   float* __restrict__ fout,
                                                   half_t* __restrict__ hout) {
    __shared__ half_t As[64][72];   // [m][k]
    __shared__ half_t BsT[64][72];  // [n][k]  (transposed B tile)
    const int tid  = threadIdx.x;
    const int wave = tid >> 5;
    const int lane = tid & 31;
    const int lane15 = lane & 15;
    const int m0 = blockIdx.y * 64;
    const int n0 = blockIdx.x * 64;

    v8f zero8 = {0.f, 0.f, 0.f, 0.f, 0.f, 0.f, 0.f, 0.f};
    v8f acc[4];
#pragma unroll
    for (int i = 0; i < 4; ++i) acc[i] = zero8;

    for (int k0 = 0; k0 < K; k0 += 64) {
        // A tile: async global->LDS 16B DMA copies (ASYNCcnt path on gfx1250)
#pragma unroll
        for (int t = 0; t < 4; ++t) {
            int chunk = tid + t * 128;           // 512 chunks of 8 halves
            int r = chunk >> 3, c8 = (chunk & 7) << 3;
            async_copy16(&A[(size_t)(m0 + r) * K + k0 + c8], &As[r][c8]);
        }
        // B tile: vector global read, transposed scalar LDS store
#pragma unroll
        for (int t = 0; t < 4; ++t) {
            int chunk = tid + t * 128;
            int r = chunk >> 3, c8 = (chunk & 7) << 3;
            v8h bv = *(const v8h*)&B[(size_t)(k0 + r) * N + n0 + c8];
#pragma unroll
            for (int i = 0; i < 8; ++i) BsT[c8 + i][r] = bv[i];
        }
        // prefetch next tiles into L2 while we compute
        if (k0 + 64 < K) {
            __builtin_prefetch(&A[(size_t)(m0 + (tid >> 1)) * K + k0 + 64], 0, 0);
            __builtin_prefetch(&B[(size_t)(k0 + 64 + (tid >> 1)) * N + n0], 0, 0);
        }
        async_wait0();
        __syncthreads();
#pragma unroll
        for (int kk = 0; kk < 64; kk += 32) {
            v16h a = frag_a(&As[0][0], 72, wave * 16, kk, lane);
#pragma unroll
            for (int nt = 0; nt < 4; ++nt) {
                v16h b = frag_bt(&BsT[0][0], 72, kk, nt * 16, lane);
                acc[nt] = wmma_f16(a, b, acc[nt]);
            }
        }
        __syncthreads();
    }

    // epilogue
    const int rbase = m0 + wave * 16 + ((lane >> 4) ? 8 : 0);
#pragma unroll
    for (int nt = 0; nt < 4; ++nt) {
        int col = n0 + nt * 16 + lane15;
        float bv = bias[col];
        int tix = 0;
        if constexpr (EPI == 0) {
            int cc = col - ((col >= 768) ? 768 : 0);
            tix = (cc / CH) * 24 + ((cc % CH) >> 1);
        }
#pragma unroll
        for (int j = 0; j < 8; ++j) {
            int row = rbase + j;
            float v = acc[nt][j] + bv;
            if constexpr (EPI == 0) {
                float partner = __shfl_xor(v, 1, 32);
                if (col < 1536) {
                    int l = row & 63;
                    float cs = ctab[l * 384 + tix];
                    float sn = ctab[24576 + l * 384 + tix];
                    v = v * cs + ((col & 1) ? partner * sn : -partner * sn);
                }
                hout[(size_t)row * N + col] = (half_t)v;
            } else if constexpr (EPI == 2) {
                v = v / (1.0f + __expf(-v));
                hout[(size_t)row * N + col] = (half_t)v;
            } else if constexpr (EPI == 1) {
                int b = row >> 12, rem = row & 4095;
                int win = rem >> 6, l = rem & 63;
                int hs  = ((win >> 3) << 3) + (l >> 3);
                int wsx = ((win & 7) << 3) + (l & 7);
                int hh = (hs + 60) & 63, ww = (wsx + 60) & 63;
                size_t dst = (((size_t)b << 12) + (hh << 6) + ww) * CDIM + col;
                float c1 = mmod[b * MODSTRIDE + 1536 + col];
                fout[dst] = (xres[dst] + c1 * v) * rsqrtf(1.0f + c1 * c1);
            } else {  // EPI == 3
                int b = row >> 12;
                float c2 = mmod[b * MODSTRIDE + 3840 + col];
                size_t dst = (size_t)row * CDIM + col;
                fout[dst] = (xres[dst] + c2 * v) * rsqrtf(1.0f + c2 * c2);
            }
        }
    }
}

// ---------------- windowed attention, one block per (head, window) ----------------
__global__ __launch_bounds__(128) void attn_kernel(const half_t* __restrict__ QKV,
                                                   half_t* __restrict__ O) {
    __shared__ half_t Qs[64][72];   // [token][ch] (ch padded 48->64 with 0)
    __shared__ half_t Ks[64][72];   // [token][ch] == (K^T)[n][k] layout for frag_bt
    __shared__ half_t VsT[64][72];  // [ch][token]
    __shared__ half_t Ps[64][72];   // softmax probs [token][token]
    const int head = blockIdx.x;
    const int wid  = blockIdx.y;
    const int tid  = threadIdx.x;
    const int wave = tid >> 5;
    const int lane = tid & 31;
    const int lane15 = lane & 15;

    const half_t* base = QKV + (size_t)wid * 64 * 2304;
    const int qoff = head * CH, koff = 768 + head * CH, voff = 1536 + head * CH;

    // zero the padded regions (ch 48..63)
    for (int i = tid; i < 64 * 16; i += 128) {
        int r = i >> 4, c = 48 + (i & 15);
        Qs[r][c] = (half_t)0.0f;
        Ks[r][c] = (half_t)0.0f;
        VsT[c][r] = (half_t)0.0f;
    }
    // Q,K: 48 halves per row = 6 x 16B async chunks
    for (int i = tid; i < 64 * 6; i += 128) {
        int r = i / 6, c8 = (i % 6) * 8;
        const half_t* rowp = base + (size_t)r * 2304;
        async_copy16(rowp + qoff + c8, &Qs[r][c8]);
        async_copy16(rowp + koff + c8, &Ks[r][c8]);
    }
    // V: scalar transpose into [ch][token]
    for (int i = tid; i < 64 * CH; i += 128) {
        int r = i / CH, c = i - r * CH;
        VsT[c][r] = base[(size_t)r * 2304 + voff + c];
    }
    async_wait0();
    __syncthreads();

    v8f zero8 = {0.f, 0.f, 0.f, 0.f, 0.f, 0.f, 0.f, 0.f};
    v8f s[4];
#pragma unroll
    for (int i = 0; i < 4; ++i) s[i] = zero8;
#pragma unroll
    for (int kk = 0; kk < 64; kk += 32) {
        v16h a = frag_a(&Qs[0][0], 72, wave * 16, kk, lane);
#pragma unroll
        for (int nt = 0; nt < 4; ++nt) {
            v16h b = frag_bt(&Ks[0][0], 72, kk, nt * 16, lane);  // B = K^T
            s[nt] = wmma_f16(a, b, s[nt]);
        }
    }

    // softmax over 64 columns; each (lane-group, j) owns one full row
    const float scale = 0.14433756729740646f;  // 1/sqrt(48)
    const int rloc = wave * 16 + ((lane >> 4) ? 8 : 0);
#pragma unroll
    for (int j = 0; j < 8; ++j) {
        float mx = s[0][j];
#pragma unroll
        for (int nt = 1; nt < 4; ++nt) mx = fmaxf(mx, s[nt][j]);
#pragma unroll
        for (int d = 1; d < 16; d <<= 1) mx = fmaxf(mx, __shfl_xor(mx, d, 32));
        float sum = 0.f;
#pragma unroll
        for (int nt = 0; nt < 4; ++nt) {
            float e = __expf((s[nt][j] - mx) * scale);
            s[nt][j] = e;
            sum += e;
        }
#pragma unroll
        for (int d = 1; d < 16; d <<= 1) sum += __shfl_xor(sum, d, 32);
        float inv = 1.0f / sum;
#pragma unroll
        for (int nt = 0; nt < 4; ++nt) Ps[rloc + j][nt * 16 + lane15] = (half_t)(s[nt][j] * inv);
    }
    __syncthreads();

    v8f o[4];
#pragma unroll
    for (int i = 0; i < 4; ++i) o[i] = zero8;
#pragma unroll
    for (int kk = 0; kk < 64; kk += 32) {
        v16h a = frag_a(&Ps[0][0], 72, wave * 16, kk, lane);
#pragma unroll
        for (int nt = 0; nt < 4; ++nt) {
            v16h b = frag_bt(&VsT[0][0], 72, kk, nt * 16, lane);  // B[k][n] = V[tok][ch]
            o[nt] = wmma_f16(a, b, o[nt]);
        }
    }
#pragma unroll
    for (int nt = 0; nt < 3; ++nt) {  // cols 48..63 are zero padding
        int col = nt * 16 + lane15;
        if (col < CH) {
#pragma unroll
            for (int j = 0; j < 8; ++j) {
                int row = rloc + j;
                O[((size_t)wid * 64 + row) * CDIM + head * CH + col] = (half_t)o[nt][j];
            }
        }
    }
}

// ---------------- launcher ----------------
extern "C" void kernel_launch(void* const* d_in, const int* in_sizes, int n_in,
                              void* d_out, int out_size, void* d_ws, size_t ws_size,
                              hipStream_t stream) {
    (void)in_sizes; (void)n_in; (void)out_size; (void)ws_size;
    const float* x      = (const float*)d_in[0];
    const float* mod    = (const float*)d_in[1];
    const float* ada_w1 = (const float*)d_in[2];
    const float* ada_b1 = (const float*)d_in[3];
    const float* ada_w2 = (const float*)d_in[4];
    const float* ada_b2 = (const float*)d_in[5];
    const float* theta  = (const float*)d_in[6];
    const float* qkv_w  = (const float*)d_in[7];
    const float* qkv_b  = (const float*)d_in[8];
    const float* proj_w = (const float*)d_in[9];
    const float* proj_b = (const float*)d_in[10];
    const float* mlp_w1 = (const float*)d_in[11];
    const float* mlp_b1 = (const float*)d_in[12];
    const float* mlp_w2 = (const float*)d_in[13];
    const float* mlp_b2 = (const float*)d_in[14];
    float* out = (float*)d_out;

    char* ws = (char*)d_ws;
    float*  m_mod = (float*)(ws + OFF_M);
    float*  ctab  = (float*)(ws + OFF_CTAB);
    half_t* wq    = (half_t*)(ws + OFF_WQ);
    half_t* wp    = (half_t*)(ws + OFF_WP);
    half_t* w1h   = (half_t*)(ws + OFF_W1);
    half_t* w2h   = (half_t*)(ws + OFF_W2);
    half_t* Y     = (half_t*)(ws + OFF_Y);     // LN1 output (window order); reused as Z
    half_t* QKVb  = (half_t*)(ws + OFF_QKV);   // reused as MLP hidden H
    half_t* Ob    = (half_t*)(ws + OFF_O);
    float*  y1    = (float*)(ws + OFF_Y1);
    half_t* Z = Y;
    half_t* H = QKVb;

    // weight conversions f32 -> f16
    f32_to_f16_kernel<<<(768 * 2304 + 255) / 256, 256, 0, stream>>>(qkv_w, wq, 768 * 2304);
    f32_to_f16_kernel<<<(768 * 768  + 255) / 256, 256, 0, stream>>>(proj_w, wp, 768 * 768);
    f32_to_f16_kernel<<<(768 * 3072 + 255) / 256, 256, 0, stream>>>(mlp_w1, w1h, 768 * 3072);
    f32_to_f16_kernel<<<(3072 * 768 + 255) / 256, 256, 0, stream>>>(mlp_w2, w2h, 3072 * 768);

    // modulation + rope tables
    ada_kernel<<<8, 256, 0, stream>>>(mod, ada_w1, ada_b1, ada_w2, ada_b2, m_mod);
    rope_table_kernel<<<96, 256, 0, stream>>>(theta, ctab);

    // LN1 + modulation + shift + window partition
    ln_mod_kernel<0><<<MROWS, 256, 0, stream>>>(x, m_mod, Y);

    // QKV projection (+bias +RoPE)
    gemm_kernel<0><<<dim3(2304 / 64, MROWS / 64), 128, 0, stream>>>(
        Y, wq, qkv_b, MROWS, 2304, 768, m_mod, ctab, nullptr, nullptr, QKVb);

    // windowed attention
    attn_kernel<<<dim3(HEADS, 512), 128, 0, stream>>>(QKVb, Ob);

    // output projection + un-shift + c1 residual -> y1 (f32)
    gemm_kernel<1><<<dim3(768 / 64, MROWS / 64), 128, 0, stream>>>(
        Ob, wp, proj_b, MROWS, 768, 768, m_mod, ctab, x, y1, nullptr);

    // LN2 + modulation
    ln_mod_kernel<1><<<MROWS, 256, 0, stream>>>(y1, m_mod, Z);

    // MLP up + SiLU
    gemm_kernel<2><<<dim3(3072 / 64, MROWS / 64), 128, 0, stream>>>(
        Z, w1h, mlp_b1, MROWS, 3072, 768, m_mod, ctab, nullptr, nullptr, H);

    // MLP down + c2 residual vs original x -> d_out
    gemm_kernel<3><<<dim3(768 / 64, MROWS / 64), 128, 0, stream>>>(
        H, w2h, mlp_b2, MROWS, 768, 3072, m_mod, ctab, x, out, nullptr);
}
